// PraxisAttention_75960791597801
// MI455X (gfx1250) — compile-verified, matrix operand order checked by reading, not run on hardware
//
#include <hip/hip_runtime.h>

#define BATCH 2
#define SEQ   1024
#define NH    16
#define HD    64
#define HID   1024
#define NEGV  -1.0e9f

typedef __attribute__((ext_vector_type(16))) __bf16 bf16x16;
typedef __attribute__((ext_vector_type(8)))  float  f32x8;

// ---------- helpers ----------
__device__ inline unsigned short f2bf(float f) {            // RNE f32 -> bf16
    unsigned u = __float_as_uint(f);
    u += 0x7FFFu + ((u >> 16) & 1u);
    return (unsigned short)(u >> 16);
}

__device__ inline f32x8 wmma_bf16(bf16x16 a, bf16x16 b, f32x8 c) {
    // D = A(16x32) * B(32x16) + C, f32 accumulate
    return __builtin_amdgcn_wmma_f32_16x16x32_bf16(false, a, false, b,
                                                   (short)0, c, false, false);
}

// ---- async global->LDS staging (gfx1250 GLOBAL_LOAD_ASYNC_TO_LDS_B128) ----
#if defined(__has_builtin)
#if __has_builtin(__builtin_amdgcn_global_load_async_to_lds_b128)
#define HAS_ASYNC_LDS 1
#endif
#endif
#ifndef HAS_ASYNC_LDS
#define HAS_ASYNC_LDS 0
#endif

#if HAS_ASYNC_LDS
// builtin params are v4i* in addrspace(1)/(3) per hipcc diagnostics
typedef int v4i __attribute__((vector_size(16)));
typedef __attribute__((address_space(1))) v4i gv4i;
typedef __attribute__((address_space(3))) v4i lv4i;
__device__ inline void stage16(const unsigned short* g, unsigned short* l) {
    __builtin_amdgcn_global_load_async_to_lds_b128((gv4i*)g, (lv4i*)l, 0, 0);
}
__device__ inline void stage_wait() {
#if __has_builtin(__builtin_amdgcn_s_wait_asynccnt)
    __builtin_amdgcn_s_wait_asynccnt(0);
#else
    asm volatile("s_wait_asynccnt 0" ::: "memory");
#endif
}
#else
__device__ inline void stage16(const unsigned short* g, unsigned short* l) {
    *(uint4*)l = *(const uint4*)g;
}
__device__ inline void stage_wait() {}
#endif

// Load a 16x32 bf16 fragment (A layout; B layout is the mirrored case with
// row == N index).  Per ISA 7.12.2: lane<16 holds K 0..7 / 16..23, lane>=16
// holds K 8..15 / 24..31, pairs packed little-endian -> two uint4 loads.
__device__ inline bf16x16 ld_frag(const unsigned short* base, int stride) {
    const int lane = threadIdx.x & 31;
    const unsigned short* p = base + (lane & 15) * stride + ((lane >> 4) << 3);
    union { uint4 u[2]; bf16x16 v; } t;
    t.u[0] = *(const uint4*)p;
    t.u[1] = *(const uint4*)(p + 16);
    return t.v;
}

// ---------- f32 -> bf16 convert ----------
__global__ __launch_bounds__(256) void cvt_bf16(const float* __restrict__ x,
                                                unsigned short* __restrict__ y, int n) {
    int i = blockIdx.x * 256 + threadIdx.x;
    if (i < n) y[i] = f2bf(x[i]);
}

// ---------- tiled WMMA GEMM:  C[M=2048,N=1024] = A * W, K=1024 ----------
// HEADS=1: store bf16 into [B,H,L,D];  HEADS=0: store f32 row-major.
template <int HEADS>
__global__ __launch_bounds__(256) void gemm_bf16(const unsigned short* __restrict__ A,
                                                 const unsigned short* __restrict__ W,
                                                 float* __restrict__ outF,
                                                 unsigned short* __restrict__ outH) {
    __shared__ __align__(16) unsigned short As[128][40];   // [m][k], pad to 40
    __shared__ __align__(16) unsigned short Bs[128][40];   // [n][k] (transposed W)

    const int tid  = threadIdx.x;
    const int lane = tid & 31;
    const int wid  = tid >> 5;                 // 8 waves
    const int n0   = blockIdx.x * 128;
    const int m0   = blockIdx.y * 128;

    f32x8 acc[8];
    for (int j = 0; j < 8; ++j)
        for (int e = 0; e < 8; ++e) acc[j][e] = 0.f;

    for (int kc = 0; kc < HID; kc += 32) {
        // stage A: 128 rows x 32 k (async b128 if available)
        for (int c = tid; c < 512; c += 256) {
            int row = c >> 2, kk = (c & 3) * 8;
            stage16(&A[(size_t)(m0 + row) * HID + kc + kk], &As[row][kk]);
        }
        // stage W transposed: Bs[n][k] = W[kc+k][n0+n]
        for (int i = tid; i < 4096; i += 256) {
            int n = i >> 5, k = i & 31;
            Bs[n][k] = W[(size_t)(kc + k) * HID + n0 + n];
        }
        stage_wait();
        __syncthreads();

        bf16x16 af = ld_frag(&As[wid * 16][0], 40);
        for (int j = 0; j < 8; ++j) {
            bf16x16 bfm = ld_frag(&Bs[j * 16][0], 40);
            acc[j] = wmma_bf16(af, bfm, acc[j]);
        }
        __syncthreads();
    }

    const int ncol = lane & 15, half8 = (lane >> 4) << 3;
    for (int j = 0; j < 8; ++j)
        for (int e = 0; e < 8; ++e) {
            int m = m0 + wid * 16 + e + half8;
            int n = n0 + j * 16 + ncol;
            float v = acc[j][e];
            if (HEADS) {
                int b = m >> 10, l = m & 1023, h = n >> 6, d = n & 63;
                outH[(((size_t)(b * NH + h)) * SEQ + l) * HD + d] = f2bf(v);
            } else {
                outF[(size_t)m * HID + n] = v;
            }
        }
}

// ---------- online softmax update for one branch ----------
__device__ inline void online_update(f32x8 (&s)[4], f32x8& mr, f32x8& lr,
                                     f32x8 (&oa)[4], unsigned short* pb,
                                     int ncol, int half8) {
    f32x8 mt;
    for (int e = 0; e < 8; ++e)
        mt[e] = fmaxf(fmaxf(s[0][e], s[1][e]), fmaxf(s[2][e], s[3][e]));
    for (int off = 1; off < 16; off <<= 1)
        for (int e = 0; e < 8; ++e)
            mt[e] = fmaxf(mt[e], __shfl_xor(mt[e], off, 32));

    f32x8 mn, al, rs;
    for (int e = 0; e < 8; ++e) {
        mn[e] = fmaxf(mr[e], mt[e]);
        al[e] = __expf(mr[e] - mn[e]);
        rs[e] = 0.f;
    }
    for (int j = 0; j < 4; ++j)
        for (int e = 0; e < 8; ++e) {
            float p = __expf(s[j][e] - mn[e]);
            rs[e] += p;
            pb[(e + half8) * 72 + j * 16 + ncol] = f2bf(p);
        }
    for (int off = 1; off < 16; off <<= 1)
        for (int e = 0; e < 8; ++e) rs[e] += __shfl_xor(rs[e], off, 32);
    for (int e = 0; e < 8; ++e) { lr[e] = lr[e] * al[e] + rs[e]; mr[e] = mn[e]; }
    for (int j = 0; j < 4; ++j)
        for (int e = 0; e < 8; ++e) oa[j][e] *= al[e];
}

// ---------- differential flash attention ----------
__global__ __launch_bounds__(256) void attn_kernel(
        const unsigned short* __restrict__ q0b, const unsigned short* __restrict__ q1b,
        const unsigned short* __restrict__ k0b, const unsigned short* __restrict__ k1b,
        const unsigned short* __restrict__ vb,  const float* __restrict__ amask,
        const float* __restrict__ l0w, const float* __restrict__ l1w,
        float* __restrict__ oacc) {
    __shared__ __align__(16) unsigned short k0s[64][72];
    __shared__ __align__(16) unsigned short k1s[64][72];
    __shared__ __align__(16) unsigned short vts[64][72];       // [d][kpos] (V^T)
    __shared__ __align__(16) unsigned short pbuf[8][16][72];   // per-wave P tile

    const int tid = threadIdx.x, lane = tid & 31, wid = tid >> 5;
    const int bid  = blockIdx.x;
    const int qblk = bid & 7;
    const int h    = (bid >> 3) & 15;
    const int b    = bid >> 7;
    const int qb0  = qblk * 128;
    const int ncol = lane & 15, half8 = (lane >> 4) << 3;

    const float scale = 0.125f;                       // 1/sqrt(64)
    const float slope = exp2f(-0.5f * (float)(h + 1));
    float dot = 0.f;
    for (int i = 0; i < HD; ++i) dot += l0w[h * HD + i] * l1w[h * HD + i];
    const float lam = 0.8f + __expf(dot);

    const size_t headoff = (size_t)(b * NH + h) * SEQ * HD;

    // q fragments live in registers: 16 rows x 64 d = 2 K-chunks each
    const int qrow0 = qb0 + wid * 16;
    bf16x16 q0f[2], q1f[2];
    for (int c = 0; c < 2; ++c) {
        q0f[c] = ld_frag(q0b + headoff + (size_t)qrow0 * HD + c * 32, HD);
        q1f[c] = ld_frag(q1b + headoff + (size_t)qrow0 * HD + c * 32, HD);
    }

    // separable ALiBi: -slope*(qp-kp) = slope*kp - slope*qp. Precompute per-row
    // terms once (removes per-element i32->f32 converts from the hot loop).
    f32x8 qsl, qpos;
    for (int e = 0; e < 8; ++e) {
        float qp = (float)(qrow0 + e + half8);
        qpos[e] = qp;
        qsl[e]  = slope * qp;
    }

    f32x8 o0a[4], o1a[4], m0r, l0r, m1r, l1r;
    for (int j = 0; j < 4; ++j)
        for (int e = 0; e < 8; ++e) { o0a[j][e] = 0.f; o1a[j][e] = 0.f; }
    for (int e = 0; e < 8; ++e) {
        m0r[e] = -3.0e38f; m1r[e] = -3.0e38f; l0r[e] = 0.f; l1r[e] = 0.f;
    }

    const int kend = qb0 + 128;                       // causal pruning
    for (int kt = 0; kt < kend; kt += 64) {
        // stage K0/K1 tiles [kpos][d] (async b128) and V^T tile [d][kpos]
        for (int c = tid; c < 512; c += 256) {
            int row = c >> 3, dq = (c & 7) * 8;
            stage16(&k0b[headoff + (size_t)(kt + row) * HD + dq], &k0s[row][dq]);
            stage16(&k1b[headoff + (size_t)(kt + row) * HD + dq], &k1s[row][dq]);
        }
        for (int i = tid; i < 4096; i += 256) {
            int kl = i >> 6, d = i & 63;
            vts[d][kl] = vb[headoff + (size_t)(kt + kl) * HD + d];
        }
        stage_wait();
        __syncthreads();

        // scores: s = q k^T * scale + (slope*kp - slope*qp) + causal + pad
        f32x8 s0[4], s1[4];
        for (int j = 0; j < 4; ++j) {
            f32x8 c0, c1;
            for (int e = 0; e < 8; ++e) { c0[e] = 0.f; c1[e] = 0.f; }
            for (int c = 0; c < 2; ++c) {
                bf16x16 kb0 = ld_frag(&k0s[j * 16][c * 32], 72);
                bf16x16 kb1 = ld_frag(&k1s[j * 16][c * 32], 72);
                c0 = wmma_bf16(q0f[c], kb0, c0);
                c1 = wmma_bf16(q1f[c], kb1, c1);
            }
            int kp = kt + j * 16 + ncol;
            float kpf  = (float)kp;
            float kadd = slope * kpf + (1.0f - amask[b * SEQ + kp]) * NEGV;
            for (int e = 0; e < 8; ++e) {
                float ad = kadd - qsl[e] + ((kpf > qpos[e]) ? NEGV : 0.0f);
                s0[j][e] = c0[e] * scale + ad;
                s1[j][e] = c1[e] * scale + ad;
            }
        }

        unsigned short* pb = &pbuf[wid][0][0];

        // branch 0: softmax update, then P0 @ V
        online_update(s0, m0r, l0r, o0a, pb, ncol, half8);
        asm volatile("s_wait_dscnt 0" ::: "memory");   // wave-local LDS RAW
        for (int c = 0; c < 2; ++c) {
            bf16x16 pf = ld_frag(pb + c * 32, 72);
            for (int jd = 0; jd < 4; ++jd) {
                bf16x16 vf = ld_frag(&vts[jd * 16][c * 32], 72);
                o0a[jd] = wmma_bf16(pf, vf, o0a[jd]);
            }
        }

        // branch 1: softmax update, then P1 @ V (same per-wave buffer; DS in-order)
        online_update(s1, m1r, l1r, o1a, pb, ncol, half8);
        asm volatile("s_wait_dscnt 0" ::: "memory");
        for (int c = 0; c < 2; ++c) {
            bf16x16 pf = ld_frag(pb + c * 32, 72);
            for (int jd = 0; jd < 4; ++jd) {
                bf16x16 vf = ld_frag(&vts[jd * 16][c * 32], 72);
                o1a[jd] = wmma_bf16(pf, vf, o1a[jd]);
            }
        }
        __syncthreads();
    }

    // finalize: out = lam * (o0/l0 - o1/l1)
    f32x8 inv0, inv1;
    for (int e = 0; e < 8; ++e) { inv0[e] = 1.0f / l0r[e]; inv1[e] = 1.0f / l1r[e]; }
    for (int jd = 0; jd < 4; ++jd)
        for (int e = 0; e < 8; ++e) {
            int qp = qrow0 + e + half8;
            int d  = jd * 16 + ncol;
            oacc[headoff + (size_t)qp * HD + d] =
                lam * (o0a[jd][e] * inv0[e] - o1a[jd][e] * inv1[e]);
        }
}

// ---------- GroupNorm per (b,h) over L*D + affine + (1-lambda_init) ----------
__global__ __launch_bounds__(256) void groupnorm_k(const float* __restrict__ o,
                                                   const float* __restrict__ gnw,
                                                   const float* __restrict__ gnb,
                                                   unsigned short* __restrict__ yb) {
    __shared__ float ssum[256], ssq[256];
    const int bh = blockIdx.x, b = bh >> 4, h = bh & 15;
    const float* x = o + (size_t)bh * SEQ * HD;

    float s = 0.f, q = 0.f;
    for (int i = threadIdx.x; i < SEQ * HD; i += 256) {
        float v = x[i]; s += v; q += v * v;
    }
    ssum[threadIdx.x] = s; ssq[threadIdx.x] = q;
    __syncthreads();
    for (int o2 = 128; o2 > 0; o2 >>= 1) {
        if (threadIdx.x < o2) {
            ssum[threadIdx.x] += ssum[threadIdx.x + o2];
            ssq[threadIdx.x]  += ssq[threadIdx.x + o2];
        }
        __syncthreads();
    }
    const float mean = ssum[0] * (1.f / (SEQ * HD));
    const float var  = ssq[0] * (1.f / (SEQ * HD)) - mean * mean;
    const float inv  = rsqrtf(var + 1e-5f);

    for (int i = threadIdx.x; i < SEQ * HD; i += 256) {
        int l = i >> 6, d = i & 63;
        float v = (x[i] - mean) * inv;
        v = v * gnw[h * HD + d] + gnb[h * HD + d];
        v *= 0.2f;                                    // (1 - lambda_init)
        yb[((size_t)(b * SEQ + l)) * HID + h * HD + d] = f2bf(v);
    }
}

// ---------- launcher ----------
extern "C" void kernel_launch(void* const* d_in, const int* in_sizes, int n_in,
                              void* d_out, int out_size, void* d_ws, size_t ws_size,
                              hipStream_t stream) {
    (void)in_sizes; (void)n_in; (void)out_size; (void)ws_size;
    const float* X   = (const float*)d_in[0];
    const float* msk = (const float*)d_in[1];
    const float* Wsrc[6] = { (const float*)d_in[2], (const float*)d_in[3],
                             (const float*)d_in[4], (const float*)d_in[5],
                             (const float*)d_in[6], (const float*)d_in[7] };
    const float* l0  = (const float*)d_in[8];
    const float* l1  = (const float*)d_in[9];
    const float* gnw = (const float*)d_in[10];
    const float* gnb = (const float*)d_in[11];

    char* base = (char*)d_ws;
    size_t off = 0;
    auto alloc = [&](size_t bytes) -> void* {
        void* p = base + off;
        off = (off + bytes + 255) & ~(size_t)255;
        return p;
    };
    const size_t M = (size_t)BATCH * SEQ;             // 2048
    unsigned short* Xb = (unsigned short*)alloc(M * HID * 2);
    unsigned short* Wb[6];
    for (int i = 0; i < 6; ++i) Wb[i] = (unsigned short*)alloc((size_t)HID * HID * 2);
    unsigned short* q0b = (unsigned short*)alloc(M * HID * 2);
    unsigned short* q1b = (unsigned short*)alloc(M * HID * 2);
    unsigned short* k0b = (unsigned short*)alloc(M * HID * 2);
    unsigned short* k1b = (unsigned short*)alloc(M * HID * 2);
    unsigned short* vbf = (unsigned short*)alloc(M * HID * 2);
    float*          oac = (float*)alloc(M * HID * 4);
    unsigned short* yb  = (unsigned short*)alloc(M * HID * 2);

    const int nX = (int)(M * HID);
    cvt_bf16<<<(nX + 255) / 256, 256, 0, stream>>>(X, Xb, nX);
    for (int i = 0; i < 6; ++i)
        cvt_bf16<<<(HID * HID + 255) / 256, 256, 0, stream>>>(Wsrc[i], Wb[i], HID * HID);

    dim3 gg(HID / 128, (unsigned)(M / 128));          // (8,16)
    gemm_bf16<1><<<gg, 256, 0, stream>>>(Xb, Wb[0], nullptr, q0b);
    gemm_bf16<1><<<gg, 256, 0, stream>>>(Xb, Wb[1], nullptr, q1b);
    gemm_bf16<1><<<gg, 256, 0, stream>>>(Xb, Wb[2], nullptr, k0b);
    gemm_bf16<1><<<gg, 256, 0, stream>>>(Xb, Wb[3], nullptr, k1b);
    gemm_bf16<1><<<gg, 256, 0, stream>>>(Xb, Wb[4], nullptr, vbf);

    attn_kernel<<<BATCH * NH * (SEQ / 128), 256, 0, stream>>>(
        q0b, q1b, k0b, k1b, vbf, msk, l0, l1, oac);

    groupnorm_k<<<BATCH * NH, 256, 0, stream>>>(oac, gnw, gnb, yb);

    gemm_bf16<0><<<gg, 256, 0, stream>>>(yb, Wb[5], (float*)d_out, nullptr);
}
